// EncoderDecoder_17772574671475
// MI455X (gfx1250) — matrix operand order repaired and checked by reading, not compiled, not run on hardware
//
#include <hip/hip_runtime.h>
#include <hip/hip_bf16.h>
#include <stdint.h>

#define DEV __device__ __forceinline__

typedef __attribute__((ext_vector_type(16))) __bf16 v16bf;
typedef __attribute__((ext_vector_type(8)))  __bf16 bf16x8;
typedef __attribute__((ext_vector_type(8)))  float  v8f;

namespace {
constexpr int NB = 64;      // batch
constexpr int NS = 48;      // source length
constexpr int NT = 48;      // target length
constexpr int NH = 1024;    // hidden
constexpr int NV = 32000;   // vocab
constexpr int NH4 = 4 * NH; // 4096
constexpr int NH2 = 2 * NH; // 2048
constexpr int PF  = 256;    // prefetch distance in bf16 elements (512 B)
}

union FragU { v16bf v; bf16x8 h[2]; };

// A fragment: 16x32 bf16 tile, rows m0..m0+15 of a row-major [*, ld] matrix.
// lane L<16: M=L, K = {k0+0..7, k0+16..23}; lane L>=16: M=L-16, K = {k0+8..15, k0+24..31}
DEV v16bf load_a_frag(const __bf16* base, int ld, int m0, int k0) {
  int lane = threadIdx.x & 31;
  int r = lane & 15;
  int kh = (lane >> 4) << 3; // 0 or 8
  const __bf16* p = base + (size_t)(m0 + r) * ld + k0 + kh;
  FragU u;
  u.h[0] = *(const bf16x8*)p;
  u.h[1] = *(const bf16x8*)(p + 16);
  return u.v;
}

// B fragment: 32x16 (KxN) tile of W^T, i.e. rows n0..n0+15 of row-major W[N][K].
// lane L<16: N=L, K = {k0+0..7, k0+16..23}; lane L>=16: N=L-16, K = {k0+8..15, k0+24..31}
DEV v16bf load_b_frag(const __bf16* base, int ld, int n0, int k0) {
  int lane = threadIdx.x & 31;
  int r = lane & 15;
  int kh = (lane >> 4) << 3; // 0 or 8
  const __bf16* p = base + (size_t)(n0 + r) * ld + k0 + kh;
  FragU u;
  u.h[0] = *(const bf16x8*)p;
  u.h[1] = *(const bf16x8*)(p + 16);
  return u.v;
}

DEV v8f wmma_bf16(v16bf a, v16bf b, v8f c) {
  return __builtin_amdgcn_wmma_f32_16x16x32_bf16(false, a, false, b, (short)0, c, false, false);
}

DEV float sigmoidf_(float x) { return 1.0f / (1.0f + __expf(-x)); }

DEV float wave_red_sum(float v) {
  // wave32 butterfly reduction
  for (int off = 16; off > 0; off >>= 1) v += __shfl_xor(v, off, 32);
  return v;
}
DEV float wave_red_max(float v) {
  for (int off = 16; off > 0; off >>= 1) v = fmaxf(v, __shfl_xor(v, off, 32));
  return v;
}

// ---------------- utility kernels ----------------

__global__ void k_cvt_bf16(const float* __restrict__ src, __bf16* __restrict__ dst, size_t n) {
  size_t i = (size_t)blockIdx.x * blockDim.x + threadIdx.x;
  if (i < n) dst[i] = (__bf16)src[i];
}

__global__ void k_zero_f32(float* p, size_t n) {
  size_t i = (size_t)blockIdx.x * blockDim.x + threadIdx.x;
  if (i < n) p[i] = 0.0f;
}

__global__ void k_zero_bf16(__bf16* p, size_t n) {
  size_t i = (size_t)blockIdx.x * blockDim.x + threadIdx.x;
  if (i < n) p[i] = (__bf16)0.0f;
}

__global__ void k_lengths(const int* __restrict__ source, int* __restrict__ lengths) {
  int b = threadIdx.x;
  if (b < NB) {
    int c = 0;
    for (int s = 0; s < NS; ++s) c += (source[b * NS + s] != 0) ? 1 : 0;
    lengths[b] = c;
  }
}

// gather source embeddings (forward order + per-sequence-reversed order), as bf16
__global__ void k_embed_src(const float* __restrict__ E, const int* __restrict__ source,
                            const int* __restrict__ lengths,
                            __bf16* __restrict__ xs_bf, __bf16* __restrict__ xsr_bf) {
  size_t i = (size_t)blockIdx.x * blockDim.x + threadIdx.x;
  size_t total = (size_t)NS * NB * NH;
  if (i >= total) return;
  int h = (int)(i % NH);
  size_t sb = i / NH;
  int b = (int)(sb % NB);
  int s = (int)(sb / NB);
  int tok = source[b * NS + s];
  xs_bf[i] = (__bf16)E[(size_t)tok * NH + h];
  int len = lengths[b];
  int rs = (s < len) ? (len - 1 - s) : s;
  int tok2 = source[b * NS + rs];
  xsr_bf[i] = (__bf16)E[(size_t)tok2 * NH + h];
}

// decoder input embeddings for steps 0..NT-2
__global__ void k_embed_dec(const float* __restrict__ E, const int* __restrict__ target,
                            __bf16* __restrict__ emb_bf) {
  size_t i = (size_t)blockIdx.x * blockDim.x + threadIdx.x;
  size_t total = (size_t)(NT - 1) * NB * NH;
  if (i >= total) return;
  int h = (int)(i % NH);
  size_t tb = i / NH;
  int b = (int)(tb % NB);
  int t = (int)(tb / NB);
  int tok = target[b * NT + t];
  emb_bf[i] = (__bf16)E[(size_t)tok * NH + h];
}

__global__ void k_combine(const float* __restrict__ ha, const float* __restrict__ hb,
                          const float* __restrict__ ca, const float* __restrict__ cb,
                          float* __restrict__ h, float* __restrict__ c, __bf16* __restrict__ hbf) {
  int i = blockIdx.x * blockDim.x + threadIdx.x;
  if (i < NB * NH) {
    float hv = ha[i] + hb[i];
    h[i] = hv;
    hbf[i] = (__bf16)hv;
    c[i] = ca[i] + cb[i];
  }
}

// ---------------- generic WMMA GEMM: C = A @ W^T + bias (+tanh) ----------------
// A: [M,K] bf16 row-major, W: [N,K] bf16 row-major. M % 64 == 0, N % 16 == 0, K % 32 == 0.
// Each wave handles one 16-wide column tile and 64 rows (4 accumulator tiles).
// K loop is split into a branch-free main loop with unconditional W-stream prefetch
// (global_prefetch_b8, SE scope -> GL2) and a branch-free tail, so the compiler can
// software-pipeline loads against v_wmma with partial s_wait_loadcnt.
__global__ void k_gemm(const __bf16* __restrict__ A, const __bf16* __restrict__ W,
                       const float* __restrict__ bias0, const float* __restrict__ bias1,
                       float* __restrict__ C, __bf16* __restrict__ Cbf,
                       int M, int N, int K, int act) {
  int wid = (int)((blockIdx.x * blockDim.x + threadIdx.x) >> 5);
  int ntn = N >> 4, nmg = M >> 6;
  if (wid >= nmg * ntn) return;
  int mg = wid / ntn, nt = wid % ntn;
  int n0 = nt << 4, m0 = mg << 6;
  int lane = threadIdx.x & 31;
  const __bf16* wrow = W + (size_t)(n0 + (lane & 15)) * K + ((lane >> 4) << 4);
  v8f acc[4] = {v8f{}, v8f{}, v8f{}, v8f{}};
  int kmain = (K > PF) ? (K - PF) : 0;
  int k = 0;
  for (; k < kmain; k += 32) {
    __builtin_prefetch((const void*)(wrow + k + PF), 0, 3);
    v16bf bf = load_b_frag(W, K, n0, k);
    acc[0] = wmma_bf16(load_a_frag(A, K, m0 +  0, k), bf, acc[0]);
    acc[1] = wmma_bf16(load_a_frag(A, K, m0 + 16, k), bf, acc[1]);
    acc[2] = wmma_bf16(load_a_frag(A, K, m0 + 32, k), bf, acc[2]);
    acc[3] = wmma_bf16(load_a_frag(A, K, m0 + 48, k), bf, acc[3]);
  }
  for (; k < K; k += 32) {
    v16bf bf = load_b_frag(W, K, n0, k);
    acc[0] = wmma_bf16(load_a_frag(A, K, m0 +  0, k), bf, acc[0]);
    acc[1] = wmma_bf16(load_a_frag(A, K, m0 + 16, k), bf, acc[1]);
    acc[2] = wmma_bf16(load_a_frag(A, K, m0 + 32, k), bf, acc[2]);
    acc[3] = wmma_bf16(load_a_frag(A, K, m0 + 48, k), bf, acc[3]);
  }
  int n = n0 + (lane & 15);
  int mo = (lane >> 4) << 3;
  float bsum = (bias0 ? bias0[n] : 0.0f) + (bias1 ? bias1[n] : 0.0f);
  for (int q = 0; q < 4; ++q) {
    for (int r = 0; r < 8; ++r) {
      int m = m0 + (q << 4) + mo + r;
      float val = acc[q][r] + bsum;
      if (act == 1) val = tanhf(val);
      size_t idx = (size_t)m * N + n;
      C[idx] = val;
      if (Cbf) Cbf[idx] = (__bf16)val;
    }
  }
}

// ---------------- fused LSTM step ----------------
// Z = h_in @ Whh^T (WMMA, all 4 gate blocks per wave) + Xpre (precomputed x@Wih^T + biases),
// then gate math + masked state update in-register.
// mode 0: decoder (mask: tgt[b*tstride+t] != 0), no output store
// mode 1: encoder fwd (mask: t < lengths[b]), out[t,b,:] = masked hn
// mode 2: encoder bwd, out[rev(t),b,:] += masked hn
__global__ void k_lstm_step(const __bf16* __restrict__ hbf_in, const float* __restrict__ h_in,
                            const float* __restrict__ c_in,
                            const float* __restrict__ Xpre_t,   // [NB, 4H]
                            const __bf16* __restrict__ Whh,     // [4H, H] bf16
                            float* __restrict__ h_out, float* __restrict__ c_out,
                            __bf16* __restrict__ hbf_out,
                            const int* __restrict__ lengths,
                            const int* __restrict__ tgt, int tstride,
                            int t, int mode, float* __restrict__ out) {
  int wid = (int)((blockIdx.x * blockDim.x + threadIdx.x) >> 5);
  if (wid >= 4 * (NH / 16)) return;
  int mt = wid >> 6;       // 0..3 row tile (NB=64)
  int nc = wid & 63;       // column tile within H
  int m0 = mt << 4, n0 = nc << 4;
  v8f zi = {}, zf = {}, zg = {}, zo = {};
  for (int k = 0; k < NH; k += 32) {
    v16bf a = load_a_frag(hbf_in, NH, m0, k);
    zi = wmma_bf16(a, load_b_frag(Whh, NH,          n0, k), zi);
    zf = wmma_bf16(a, load_b_frag(Whh, NH,     NH + n0, k), zf);
    zg = wmma_bf16(a, load_b_frag(Whh, NH, 2 * NH + n0, k), zg);
    zo = wmma_bf16(a, load_b_frag(Whh, NH, 3 * NH + n0, k), zo);
  }
  int lane = threadIdx.x & 31;
  int n = n0 + (lane & 15);
  int mo = (lane >> 4) << 3;
  for (int r = 0; r < 8; ++r) {
    int m = m0 + mo + r; // batch index
    const float* xp = Xpre_t + (size_t)m * NH4;
    float vi = sigmoidf_(zi[r] + xp[n]);
    float vf = sigmoidf_(zf[r] + xp[NH + n]);
    float vg = tanhf   (zg[r] + xp[2 * NH + n]);
    float vo = sigmoidf_(zo[r] + xp[3 * NH + n]);
    float cprev = c_in[m * NH + n];
    float cn = vf * cprev + vi * vg;
    float hn = vo * tanhf(cn);
    bool valid;
    if (mode == 0) valid = (tgt[m * tstride + t] != 0);
    else           valid = (t < lengths[m]);
    float h2 = valid ? hn : h_in[m * NH + n];
    float c2 = valid ? cn : cprev;
    h_out[m * NH + n] = h2;
    c_out[m * NH + n] = c2;
    hbf_out[m * NH + n] = (__bf16)h2;
    if (mode == 1) {
      out[((size_t)t * NB + m) * NH + n] = valid ? hn : 0.0f;
    } else if (mode == 2) {
      int len = lengths[m];
      int rt = (t < len) ? (len - 1 - t) : t;
      out[((size_t)rt * NB + m) * NH + n] += valid ? hn : 0.0f;
    }
  }
}

// ---------------- attention: scores -> masked softmax -> context; emit cat(ctx,h2) as bf16 ----
// One block (256 threads = 8 waves) per batch row. Wave-level shuffle reductions, then
// a tiny 8-partial combine in LDS (cuts barrier count on the serial decoder path).
__global__ void k_attention(const float* __restrict__ h2, const float* __restrict__ enc,
                            const int* __restrict__ source, __bf16* __restrict__ cat_bf) {
  int b = blockIdx.x;
  int tid = threadIdx.x; // 256
  int wv = tid >> 5, lane = tid & 31;
  __shared__ float part[8];
  __shared__ float sa[NS];
  const float* hh = h2 + (size_t)b * NH;
  for (int s = 0; s < NS; ++s) {
    const float* e = enc + ((size_t)s * NB + b) * NH;
    float p = 0.0f;
    for (int h = tid; h < NH; h += 256) p += hh[h] * e[h];
    p = wave_red_sum(p);
    if (lane == 0) part[wv] = p;
    __syncthreads();
    if (tid == 0) {
      float acc = 0.0f;
      for (int w = 0; w < 8; ++w) acc += part[w];
      sa[s] = acc;
    }
    __syncthreads();
  }
  if (tid == 0) {
    float mx = -3.4e38f;
    for (int s = 0; s < NS; ++s)
      if (source[b * NS + s] != 0 && sa[s] > mx) mx = sa[s];
    float se = 0.0f;
    for (int s = 0; s < NS; ++s) {
      float a = (source[b * NS + s] != 0) ? __expf(sa[s] - mx) : 0.0f;
      sa[s] = a; se += a;
    }
    float inv = 1.0f / se;
    for (int s = 0; s < NS; ++s) sa[s] *= inv;
  }
  __syncthreads();
  for (int h = tid; h < NH; h += 256) {
    float acc = 0.0f;
    for (int s = 0; s < NS; ++s) acc += sa[s] * enc[((size_t)s * NB + b) * NH + h];
    cat_bf[(size_t)b * NH2 + h]      = (__bf16)acc;
    cat_bf[(size_t)b * NH2 + NH + h] = (__bf16)hh[h];
  }
}

// ---------------- per-row NLL (log-softmax over V, gather at wt) ----------------
__global__ void k_rownll(const float* __restrict__ logits, const int* __restrict__ target,
                         int t, float* __restrict__ nll, float* __restrict__ msk) {
  int b = blockIdx.x, tid = threadIdx.x;
  int wv = tid >> 5, lane = tid & 31;
  __shared__ float part[8];
  __shared__ float bcast;
  const float* row = logits + (size_t)b * NV;
  float mx = -3.4e38f;
  for (int v = tid; v < NV; v += 256) mx = fmaxf(mx, row[v]);
  mx = wave_red_max(mx);
  if (lane == 0) part[wv] = mx;
  __syncthreads();
  if (tid == 0) {
    float m = part[0];
    for (int w = 1; w < 8; ++w) m = fmaxf(m, part[w]);
    bcast = m;
  }
  __syncthreads();
  mx = bcast;
  float se = 0.0f;
  for (int v = tid; v < NV; v += 256) se += __expf(row[v] - mx);
  se = wave_red_sum(se);
  if (lane == 0) part[wv] = se;
  __syncthreads();
  if (tid == 0) {
    float s = 0.0f;
    for (int w = 0; w < 8; ++w) s += part[w];
    int wt = target[b * NT + (t + 1)];
    float lse = mx + logf(s);
    float m = (wt != 0) ? 1.0f : 0.0f;
    nll[b] = -(row[wt] - lse) * m;
    msk[b] = m;
  }
}

__global__ void k_loss_acc(const float* __restrict__ nll, const float* __restrict__ msk,
                           float* __restrict__ loss) {
  __shared__ float s1[64], s2[64];
  int tid = threadIdx.x;
  s1[tid] = nll[tid]; s2[tid] = msk[tid]; __syncthreads();
  for (int off = 32; off > 0; off >>= 1) {
    if (tid < off) { s1[tid] += s1[tid + off]; s2[tid] += s2[tid + off]; }
    __syncthreads();
  }
  if (tid == 0) *loss += s1[0] / fmaxf(s2[0], 1.0f);
}

__global__ void k_finalize(const float* __restrict__ loss, float* __restrict__ out) {
  out[0] = loss[0];
}

// ---------------- host orchestration ----------------

extern "C" void kernel_launch(void* const* d_in, const int* in_sizes, int n_in,
                              void* d_out, int out_size, void* d_ws, size_t ws_size,
                              hipStream_t stream) {
  (void)in_sizes; (void)n_in; (void)out_size; (void)ws_size;
  const int*   src    = (const int*)d_in[0];
  const int*   tgt    = (const int*)d_in[1];
  const float* E_src  = (const float*)d_in[2];
  const float* Wih_f  = (const float*)d_in[3];
  const float* Whh_f  = (const float*)d_in[4];
  const float* bih_f  = (const float*)d_in[5];
  const float* bhh_f  = (const float*)d_in[6];
  const float* Wih_b  = (const float*)d_in[7];
  const float* Whh_b  = (const float*)d_in[8];
  const float* bih_b  = (const float*)d_in[9];
  const float* bhh_b  = (const float*)d_in[10];
  const float* E_tgt  = (const float*)d_in[11];
  const float* Wih_d  = (const float*)d_in[12];
  const float* Whh_d  = (const float*)d_in[13];
  const float* bih_d  = (const float*)d_in[14];
  const float* bhh_d  = (const float*)d_in[15];
  const float* W_attn = (const float*)d_in[16];
  const float* b_attn = (const float*)d_in[17];
  const float* W_out  = (const float*)d_in[18];
  const float* b_out  = (const float*)d_in[19];

  uint8_t* ws = (uint8_t*)d_ws;
  size_t cur = 0;
  auto alloc = [&](size_t bytes) -> void* {
    void* p = ws + cur;
    cur = (cur + bytes + 255) & ~(size_t)255;
    return p;
  };

  const size_t WSZ = (size_t)NH4 * NH;        // 4096*1024 elements
  __bf16* Wihf_bf  = (__bf16*)alloc(WSZ * 2);
  __bf16* Whhf_bf  = (__bf16*)alloc(WSZ * 2);
  __bf16* Wihb_bf  = (__bf16*)alloc(WSZ * 2);
  __bf16* Whhb_bf  = (__bf16*)alloc(WSZ * 2);
  __bf16* Wihd_bf  = (__bf16*)alloc(WSZ * 2);
  __bf16* Whhd_bf  = (__bf16*)alloc(WSZ * 2);
  __bf16* Wattn_bf = (__bf16*)alloc((size_t)NH * NH2 * 2);
  __bf16* Wout_bf  = (__bf16*)alloc((size_t)NV * NH * 2);
  __bf16* xs_bf    = (__bf16*)alloc((size_t)NS * NB * NH * 2);
  __bf16* xsr_bf   = (__bf16*)alloc((size_t)NS * NB * NH * 2);
  __bf16* embd_bf  = (__bf16*)alloc((size_t)(NT - 1) * NB * NH * 2);
  float*  Xpre     = (float*)alloc((size_t)NS * NB * NH4 * 4);   // reused 3x
  float*  enc      = (float*)alloc((size_t)NS * NB * NH * 4);
  float*  hA[2];  float* cA[2];  __bf16* hbA[2];
  float*  hB[2];  float* cB[2];  __bf16* hbB[2];
  for (int i = 0; i < 2; ++i) {
    hA[i]  = (float*)alloc((size_t)NB * NH * 4);
    cA[i]  = (float*)alloc((size_t)NB * NH * 4);
    hbA[i] = (__bf16*)alloc((size_t)NB * NH * 2);
    hB[i]  = (float*)alloc((size_t)NB * NH * 4);
    cB[i]  = (float*)alloc((size_t)NB * NH * 4);
    hbB[i] = (__bf16*)alloc((size_t)NB * NH * 2);
  }
  __bf16* cat_bf = (__bf16*)alloc((size_t)NB * NH2 * 2);
  float*  hx     = (float*)alloc((size_t)NB * NH * 4);
  __bf16* hx_bf  = (__bf16*)alloc((size_t)NB * NH * 2);
  float*  logits = (float*)alloc((size_t)NB * NV * 4);
  int*    lengths = (int*)alloc(NB * 4);
  float*  nll    = (float*)alloc(NB * 4);
  float*  mskv   = (float*)alloc(NB * 4);
  float*  loss   = (float*)alloc(4);

  auto gb = [](size_t n) { return (unsigned)((n + 255) / 256); };
  auto gemm_blocks = [](int M, int N) { return (unsigned)(((M / 64) * (N / 16) + 7) / 8); };

  // weight conversions fp32 -> bf16
  k_cvt_bf16<<<gb(WSZ), 256, 0, stream>>>(Wih_f, Wihf_bf, WSZ);
  k_cvt_bf16<<<gb(WSZ), 256, 0, stream>>>(Whh_f, Whhf_bf, WSZ);
  k_cvt_bf16<<<gb(WSZ), 256, 0, stream>>>(Wih_b, Wihb_bf, WSZ);
  k_cvt_bf16<<<gb(WSZ), 256, 0, stream>>>(Whh_b, Whhb_bf, WSZ);
  k_cvt_bf16<<<gb(WSZ), 256, 0, stream>>>(Wih_d, Wihd_bf, WSZ);
  k_cvt_bf16<<<gb(WSZ), 256, 0, stream>>>(Whh_d, Whhd_bf, WSZ);
  k_cvt_bf16<<<gb((size_t)NH * NH2), 256, 0, stream>>>(W_attn, Wattn_bf, (size_t)NH * NH2);
  k_cvt_bf16<<<gb((size_t)NV * NH), 256, 0, stream>>>(W_out, Wout_bf, (size_t)NV * NH);

  // lengths + embeddings
  k_lengths<<<1, 64, 0, stream>>>(src, lengths);
  k_embed_src<<<gb((size_t)NS * NB * NH), 256, 0, stream>>>(E_src, src, lengths, xs_bf, xsr_bf);
  k_embed_dec<<<gb((size_t)(NT - 1) * NB * NH), 256, 0, stream>>>(E_tgt, tgt, embd_bf);

  // zero initial states + loss
  k_zero_f32<<<gb((size_t)NB * NH), 256, 0, stream>>>(hA[0], (size_t)NB * NH);
  k_zero_f32<<<gb((size_t)NB * NH), 256, 0, stream>>>(cA[0], (size_t)NB * NH);
  k_zero_bf16<<<gb((size_t)NB * NH), 256, 0, stream>>>(hbA[0], (size_t)NB * NH);
  k_zero_f32<<<gb((size_t)NB * NH), 256, 0, stream>>>(hB[0], (size_t)NB * NH);
  k_zero_f32<<<gb((size_t)NB * NH), 256, 0, stream>>>(cB[0], (size_t)NB * NH);
  k_zero_bf16<<<gb((size_t)NB * NH), 256, 0, stream>>>(hbB[0], (size_t)NB * NH);
  k_zero_f32<<<1, 1, 0, stream>>>(loss, 1);

  // ---- encoder forward ----
  k_gemm<<<gemm_blocks(NS * NB, NH4), 256, 0, stream>>>(
      xs_bf, Wihf_bf, bih_f, bhh_f, Xpre, nullptr, NS * NB, NH4, NH, 0);
  for (int t = 0; t < NS; ++t) {
    int in = t & 1, out = in ^ 1;
    k_lstm_step<<<32, 256, 0, stream>>>(hbA[in], hA[in], cA[in],
        Xpre + (size_t)t * NB * NH4, Whhf_bf,
        hA[out], cA[out], hbA[out], lengths, nullptr, 0, t, 1, enc);
  }
  // final fwd state in index 0 (48 steps)

  // ---- encoder backward ----
  k_gemm<<<gemm_blocks(NS * NB, NH4), 256, 0, stream>>>(
      xsr_bf, Wihb_bf, bih_b, bhh_b, Xpre, nullptr, NS * NB, NH4, NH, 0);
  for (int t = 0; t < NS; ++t) {
    int in = t & 1, out = in ^ 1;
    k_lstm_step<<<32, 256, 0, stream>>>(hbB[in], hB[in], cB[in],
        Xpre + (size_t)t * NB * NH4, Whhb_bf,
        hB[out], cB[out], hbB[out], lengths, nullptr, 0, t, 2, enc);
  }
  // final bwd state in index 0

  // decoder initial state hx = hf + hb, cx = cf + cb -> slot 1 of set A
  k_combine<<<gb((size_t)NB * NH), 256, 0, stream>>>(hA[0], hB[0], cA[0], cB[0],
                                                     hA[1], cA[1], hbA[1]);

  // ---- decoder input pre-GEMM ----
  k_gemm<<<gemm_blocks((NT - 1) * NB, NH4), 256, 0, stream>>>(
      embd_bf, Wihd_bf, bih_d, bhh_d, Xpre, nullptr, (NT - 1) * NB, NH4, NH, 0);

  // ---- decoder loop ----
  for (int t = 0; t < NT - 1; ++t) {
    int in = (t & 1) ^ 1, out = t & 1; // starts at slot 1
    k_lstm_step<<<32, 256, 0, stream>>>(hbA[in], hA[in], cA[in],
        Xpre + (size_t)t * NB * NH4, Whhd_bf,
        hA[out], cA[out], hbA[out], nullptr, tgt, NT, t, 0, nullptr);
    k_attention<<<NB, 256, 0, stream>>>(hA[out], enc, src, cat_bf);
    k_gemm<<<gemm_blocks(NB, NH), 256, 0, stream>>>(
        cat_bf, Wattn_bf, b_attn, nullptr, hx, hx_bf, NB, NH, NH2, 1);
    k_gemm<<<gemm_blocks(NB, NV), 256, 0, stream>>>(
        hx_bf, Wout_bf, b_out, nullptr, logits, nullptr, NB, NV, NH, 0);
    k_rownll<<<NB, 256, 0, stream>>>(logits, tgt, t, nll, mskv);
    k_loss_acc<<<1, 64, 0, stream>>>(nll, mskv, loss);
  }

  k_finalize<<<1, 1, 0, stream>>>(loss, (float*)d_out);
}
